// VAE_l1_diag_12610023981355
// MI455X (gfx1250) — matrix operand order repaired and checked by reading, not compiled, not run on hardware
//
#include <hip/hip_runtime.h>
#include <stdint.h>
#include <stddef.h>

// ---------------------------------------------------------------------------
// VAE forward for MI455X (gfx1250): bf16 WMMA GEMMs with double-buffered
// async global->LDS tile loads (ASYNCcnt), statically unrolled 2x so buffer
// selection is compile-time (no cndmask/v_nop hazards in the hot loop).
// Deterministic BN stats. Requires ws_size >= ~129 MB.
// ---------------------------------------------------------------------------

typedef __attribute__((ext_vector_type(16))) __bf16 v16bf;
typedef __attribute__((ext_vector_type(8)))  float  v8f;

union FragAB { v16bf v; uint4 u[2]; };

__device__ __forceinline__ unsigned short f2bf(float f) {
  union { float f; unsigned u; } c; c.f = f;
  unsigned u = c.u;
  unsigned r = u + 0x7FFFu + ((u >> 16) & 1u);   // round-to-nearest-even
  return (unsigned short)(r >> 16);
}

__device__ __forceinline__ unsigned pcg_hash(unsigned v) {
  unsigned s = v * 747796405u + 2891336453u;
  unsigned w = ((s >> ((s >> 28) + 4u)) ^ s) * 277803737u;
  return (w >> 22) ^ w;
}

__device__ __forceinline__ float gauss_hash(unsigned idx) {
  unsigned a = pcg_hash(idx * 2u + 1u);
  unsigned b = pcg_hash(idx * 2u + 2u);
  float u1 = ((float)a + 1.0f) * (1.0f / 4294967296.0f);
  float u2 = ((float)b) * (1.0f / 4294967296.0f);
  return sqrtf(-2.0f * logf(u1)) * cosf(6.28318530718f * u2);
}

// lds byte offset (wave-relative) from a generic shared pointer:
// flat LDS addresses keep the logical LDS offset in addr[31:0].
__device__ __forceinline__ unsigned lds_off(const void* p) {
  return (unsigned)(uintptr_t)p;
}

// async global->LDS 16-byte load, tracked by ASYNCcnt
__device__ __forceinline__ void async_load_b128(unsigned lds_addr, const void* gaddr) {
  asm volatile("global_load_async_to_lds_b128 %0, %1, off"
               :: "v"(lds_addr), "v"(gaddr) : "memory");
}
__device__ __forceinline__ void wait_async0() {
  asm volatile("s_wait_asynccnt 0x0" ::: "memory");
}

// ---------------- elementwise / conversion kernels ----------------

// out_bf16[i*N+j] = bf16(x[i*N+j] * diag[j])
__global__ __launch_bounds__(256) void scale_cvt_kernel(
    const float* __restrict__ x, const float* __restrict__ diag,
    unsigned short* __restrict__ out, int N, long long total) {
  long long idx = (long long)blockIdx.x * 256 + threadIdx.x;
  if (idx >= total) return;
  int j = (int)(idx % N);
  out[idx] = f2bf(x[idx] * diag[j]);
}

// LDS-tiled transpose-convert: W[K][N] f32 -> Wt[Npad][Kpad] bf16, zero padded.
// grid = (Kpad/32, Npad/32), 256 threads.
__global__ __launch_bounds__(256) void wtcvt_tiled_kernel(
    const float* __restrict__ W, unsigned short* __restrict__ Wt,
    int K, int N, int Kpad) {
  __shared__ float tile[32][33];
  const int tkb = blockIdx.x * 32;   // k tile base
  const int tnb = blockIdx.y * 32;   // n tile base
  const int tx = threadIdx.x & 31;
  const int ty = threadIdx.x >> 5;   // 0..7
#pragma unroll
  for (int i = 0; i < 4; ++i) {
    int k = tkb + ty + i * 8;
    int n = tnb + tx;
    tile[ty + i * 8][tx] = (k < K && n < N) ? W[(size_t)k * N + n] : 0.0f;
  }
  __syncthreads();
#pragma unroll
  for (int i = 0; i < 4; ++i) {
    int n = tnb + ty + i * 8;
    int k = tkb + tx;
    Wt[(size_t)n * Kpad + k] = f2bf(tile[tx][ty + i * 8]);
  }
}

// ---------------- WMMA GEMM ----------------
// C[M][ldc] = A_bf16[M][lda] * Wt_bf16[N][ldb]^T + bias, optional LeakyReLU.
// Block tile 128x128, 8 waves (wave32): wave tile 32x64 = 2x4 WMMA 16x16 tiles.
// K must be a multiple of 64; Wt allocated/zeroed up to gridDim.x*128 rows.
// Tiles staged via double-buffered global_load_async_to_lds_b128 with static
// buffer indices (k-loop unrolled 2x).
__global__ __launch_bounds__(256) void gemm_bf16_wmma(
    const unsigned short* __restrict__ A, const unsigned short* __restrict__ Wt,
    const float* __restrict__ bias, float* __restrict__ C,
    int N, int K, int lda, int ldb, int ldc, int act) {
  __shared__ alignas(16) unsigned short As[2][128 * 40];
  __shared__ alignas(16) unsigned short Bs[2][128 * 40];

  const int tid  = threadIdx.x;
  const int lane = tid & 31;
  const int wid  = tid >> 5;
  const int wm   = wid & 3;        // 4 waves along M -> 32 rows each
  const int wn   = wid >> 2;       // 2 waves along N -> 64 cols each
  const int half = lane >> 4;
  const int l16  = lane & 15;
  const int mBase = blockIdx.y * 128;
  const int nBase = blockIdx.x * 128;

  // per-thread cooperative load coordinates (2 rows x 8 bf16 per tile each)
  const int r0 = tid >> 2,          c0 = (tid & 3) * 8;
  const int r1 = (tid + 256) >> 2,  c1 = ((tid + 256) & 3) * 8;

  const unsigned short* gA0 = A  + (size_t)(mBase + r0) * lda + c0;
  const unsigned short* gA1 = A  + (size_t)(mBase + r1) * lda + c1;
  const unsigned short* gB0 = Wt + (size_t)(nBase + r0) * ldb + c0;
  const unsigned short* gB1 = Wt + (size_t)(nBase + r1) * ldb + c1;

  unsigned lA0[2], lA1[2], lB0[2], lB1[2];
#pragma unroll
  for (int b = 0; b < 2; ++b) {
    lA0[b] = lds_off(&As[b][r0 * 40 + c0]);
    lA1[b] = lds_off(&As[b][r1 * 40 + c1]);
    lB0[b] = lds_off(&Bs[b][r0 * 40 + c0]);
    lB1[b] = lds_off(&Bs[b][r1 * 40 + c1]);
  }

  v8f acc[2][4];
#pragma unroll
  for (int mi = 0; mi < 2; ++mi)
#pragma unroll
    for (int ni = 0; ni < 4; ++ni)
      acc[mi][ni] = (v8f){0.f, 0.f, 0.f, 0.f, 0.f, 0.f, 0.f, 0.f};

  // stage a 32-wide k-chunk into buffer b
  auto stage = [&](int b, int k0) {
    async_load_b128(lA0[b], gA0 + k0);
    async_load_b128(lA1[b], gA1 + k0);
    async_load_b128(lB0[b], gB0 + k0);
    async_load_b128(lB1[b], gB1 + k0);
  };

  // consume buffer b: load frags per ISA layouts, 8 WMMAs
  auto compute = [&](int b) {
    const unsigned short* as = As[b];
    const unsigned short* bs = Bs[b];
    FragAB afrag[2], bfrag[4];
#pragma unroll
    for (int mi = 0; mi < 2; ++mi) {
      int row = wm * 32 + mi * 16 + l16;
      // ISA 16-bit A layout: elems 0..7 = K(half*8..), 8..15 = K(16+half*8..)
      afrag[mi].u[0] = *(const uint4*)&as[row * 40 + half * 8];
      afrag[mi].u[1] = *(const uint4*)&as[row * 40 + 16 + half * 8];
    }
#pragma unroll
    for (int ni = 0; ni < 4; ++ni) {
      int col = wn * 64 + ni * 16 + l16;
      // ISA 16-bit B layout: elems 0..7 = K(half*8..), 8..15 = K(16+half*8..)
      bfrag[ni].u[0] = *(const uint4*)&bs[col * 40 + half * 8];
      bfrag[ni].u[1] = *(const uint4*)&bs[col * 40 + 16 + half * 8];
    }
#pragma unroll
    for (int mi = 0; mi < 2; ++mi)
#pragma unroll
      for (int ni = 0; ni < 4; ++ni)
        acc[mi][ni] = __builtin_amdgcn_wmma_f32_16x16x32_bf16(
            false, afrag[mi].v, false, bfrag[ni].v,
            (short)0, acc[mi][ni], false, false);
  };

  // prologue: stage k=0 into buffer 0
  stage(0, 0);

  // main loop, 2x unrolled so buffer indices are compile-time constants.
  // invariants at each wait: exactly the 4 loads of the buffer to be read are
  // outstanding; each barrier follows the last reads of the buffer about to be
  // overwritten by the subsequent stage().
  for (int k0 = 0; k0 < K; k0 += 64) {
    wait_async0();
    __syncthreads();
    stage(1, k0 + 32);          // K % 64 == 0 -> k0+32 < K always
    compute(0);

    wait_async0();
    __syncthreads();
    if (k0 + 64 < K) stage(0, k0 + 64);
    compute(1);
  }

  // epilogue: bias (+ optional LeakyReLU); C/D layout row = r + 8*half
#pragma unroll
  for (int mi = 0; mi < 2; ++mi) {
#pragma unroll
    for (int ni = 0; ni < 4; ++ni) {
      int col = nBase + wn * 64 + ni * 16 + l16;
      if (col < N) {
        float bv = bias[col];
#pragma unroll
        for (int r = 0; r < 8; ++r) {
          int row = mBase + wm * 32 + mi * 16 + half * 8 + r;
          float v = acc[mi][ni][r] + bv;
          if (act) v = (v >= 0.0f) ? v : 0.01f * v;
          C[(size_t)row * ldc + col] = v;
        }
      }
    }
  }
}

// ---------------- BatchNorm stats (deterministic two-stage) ----------------

__global__ __launch_bounds__(256) void colstats_kernel(
    const float* __restrict__ X, float* __restrict__ ps, float* __restrict__ pq,
    int H, int rowsPer) {
  int col = blockIdx.x * 256 + threadIdx.x;
  int chunk = blockIdx.y;
  const float* p = X + (size_t)chunk * rowsPer * H + col;
  float s = 0.f, q = 0.f;
  for (int r = 0; r < rowsPer; ++r) {
    float v = p[(size_t)r * H];
    s += v; q += v * v;
  }
  ps[(size_t)chunk * H + col] = s;
  pq[(size_t)chunk * H + col] = q;
}

__global__ __launch_bounds__(256) void bnfinal_kernel(
    const float* __restrict__ ps, const float* __restrict__ pq,
    const float* __restrict__ g, const float* __restrict__ b,
    float* __restrict__ scale, float* __restrict__ shift,
    int H, int chunks, float invB) {
  int c = blockIdx.x * 256 + threadIdx.x;
  if (c >= H) return;
  float s = 0.f, q = 0.f;
  for (int i = 0; i < chunks; ++i) { s += ps[(size_t)i * H + c]; q += pq[(size_t)i * H + c]; }
  float m = s * invB;
  float var = q * invB - m * m;
  float sc = g[c] * rsqrtf(var + 1e-5f);
  scale[c] = sc;
  shift[c] = b[c] - m * sc;
}

__global__ __launch_bounds__(256) void bnlrelu_cvt_kernel(
    const float* __restrict__ X, const float* __restrict__ scale,
    const float* __restrict__ shift, unsigned short* __restrict__ out,
    int H, long long total) {
  long long idx = (long long)blockIdx.x * 256 + threadIdx.x;
  if (idx >= total) return;
  int c = (int)(idx % H);
  float v = X[idx] * scale[c] + shift[c];
  v = (v >= 0.0f) ? v : 0.01f * v;
  out[idx] = f2bf(v);
}

// z = mu + eps*exp(0.5*logvar), packed into bf16 [B][64] (cols 40..63 zero)
__global__ __launch_bounds__(256) void reparam_kernel(
    const float* __restrict__ muL, const float* __restrict__ lvL,
    unsigned short* __restrict__ zout, int Z, int Zpad, long long total) {
  long long idx = (long long)blockIdx.x * 256 + threadIdx.x;
  if (idx >= total) return;
  int j = (int)(idx % Zpad);
  long long i = idx / Zpad;
  float z = 0.0f;
  if (j < Z) {
    float mu = muL[(size_t)i * Z + j];
    float lv = lvL[(size_t)i * Z + j];
    float stdv = expf(0.5f * lv);
    z = mu + gauss_hash((unsigned)(i * Z + j)) * stdv;
  }
  zout[idx] = f2bf(z);
}

// ---------------------------------------------------------------------------

extern "C" void kernel_launch(void* const* d_in, const int* in_sizes, int n_in,
                              void* d_out, int out_size, void* d_ws, size_t ws_size,
                              hipStream_t stream) {
  const int B = 4096, N = 8192, H = 2048, Z = 40;
  const int Zpad = 64, ZNpad = 128;  // K-pad / N-pad for the Z-sized GEMMs

  const float* x      = (const float*)d_in[0];
  const float* diag   = (const float*)d_in[1];
  const float* enc_w1 = (const float*)d_in[2];
  const float* enc_b1 = (const float*)d_in[3];
  const float* bn1_g  = (const float*)d_in[4];
  const float* bn1_b  = (const float*)d_in[5];
  const float* enc_w2 = (const float*)d_in[6];
  const float* enc_b2 = (const float*)d_in[7];
  const float* bn2_g  = (const float*)d_in[8];
  const float* bn2_b  = (const float*)d_in[9];
  const float* enc_w3 = (const float*)d_in[10];
  const float* enc_b3 = (const float*)d_in[11];
  const float* bn3_g  = (const float*)d_in[12];
  const float* bn3_b  = (const float*)d_in[13];
  const float* w21    = (const float*)d_in[14];
  const float* b21    = (const float*)d_in[15];
  const float* w22    = (const float*)d_in[16];
  const float* b22    = (const float*)d_in[17];
  const float* w3     = (const float*)d_in[18];
  const float* b3     = (const float*)d_in[19];
  const float* bnd_g  = (const float*)d_in[20];
  const float* bnd_b  = (const float*)d_in[21];
  const float* w4     = (const float*)d_in[22];
  const float* b4     = (const float*)d_in[23];
  const float* w5     = (const float*)d_in[24];
  const float* b5     = (const float*)d_in[25];

  float* out      = (float*)d_out;
  float* out_mu_x = out;                                 // [B][N]
  float* out_lv_x = out + (size_t)B * N;                 // [B][N]
  float* out_mu_l = out + 2 * (size_t)B * N;             // [B][Z]
  float* out_lv_l = out_mu_l + (size_t)B * Z;            // [B][Z]

  // workspace layout
  char* ws = (char*)d_ws;
  unsigned short* WBUF = (unsigned short*)(ws);                        // 32 MB bf16 weight [Npad][Kpad]
  unsigned short* ABUF = (unsigned short*)(ws + (32ll << 20));         // 64 MB bf16 activation
  float*          FBUF = (float*)(ws + (96ll << 20));                  // 32 MB f32 GEMM out [B][H]
  float*          PS   = (float*)(ws + (128ll << 20));                 // 32*H partial sums
  float*          PQ   = PS + 32 * H;
  float*          SC   = PQ + 32 * H;
  float*          SH   = SC + H;
  (void)ws_size; (void)in_sizes; (void)n_in; (void)out_size;

  const int chunks = 32, rowsPer = B / chunks;  // 128
  const float invB = 1.0f / (float)B;

  // ---- h0 = x * diag -> bf16 [B][N] ----
  {
    long long total = (long long)B * N;
    scale_cvt_kernel<<<(unsigned)((total + 255) / 256), 256, 0, stream>>>(
        x, diag, ABUF, N, total);
  }

  // one Linear + BN + LeakyReLU layer (input bf16 in ABUF, output bf16 back in ABUF)
  auto run_layer = [&](const float* W, const float* bias, const float* g,
                       const float* bb, int K, int Kpad, int Nout) {
    dim3 tgrid(Kpad / 32, Nout / 32);
    wtcvt_tiled_kernel<<<tgrid, 256, 0, stream>>>(W, WBUF, K, Nout, Kpad);
    dim3 grid(Nout / 128, B / 128);
    gemm_bf16_wmma<<<grid, 256, 0, stream>>>(
        ABUF, WBUF, bias, FBUF, Nout, Kpad, Kpad, Kpad, Nout, 0);
    dim3 sgrid(Nout / 256, chunks);
    colstats_kernel<<<sgrid, 256, 0, stream>>>(FBUF, PS, PQ, Nout, rowsPer);
    bnfinal_kernel<<<(Nout + 255) / 256, 256, 0, stream>>>(
        PS, PQ, g, bb, SC, SH, Nout, chunks, invB);
    long long total = (long long)B * Nout;
    bnlrelu_cvt_kernel<<<(unsigned)((total + 255) / 256), 256, 0, stream>>>(
        FBUF, SC, SH, ABUF, Nout, total);
  };

  // ---- encoder ----
  run_layer(enc_w1, enc_b1, bn1_g, bn1_b, N, N, H);   // [B,N] -> [B,H]
  run_layer(enc_w2, enc_b2, bn2_g, bn2_b, H, H, H);
  run_layer(enc_w3, enc_b3, bn3_g, bn3_b, H, H, H);   // h1 bf16 in ABUF

  // ---- heads: mu_latent / logvar_latent (N=40 padded to 128) ----
  {
    dim3 tgrid(H / 32, ZNpad / 32);
    dim3 grid(ZNpad / 128, B / 128);
    wtcvt_tiled_kernel<<<tgrid, 256, 0, stream>>>(w21, WBUF, H, Z, H);
    gemm_bf16_wmma<<<grid, 256, 0, stream>>>(
        ABUF, WBUF, b21, out_mu_l, Z, H, H, H, Z, 0);
    wtcvt_tiled_kernel<<<tgrid, 256, 0, stream>>>(w22, WBUF, H, Z, H);
    gemm_bf16_wmma<<<grid, 256, 0, stream>>>(
        ABUF, WBUF, b22, out_lv_l, Z, H, H, H, Z, 0);
  }

  // ---- reparameterize: z bf16 [B][64] into ABUF ----
  {
    long long total = (long long)B * Zpad;
    reparam_kernel<<<(unsigned)((total + 255) / 256), 256, 0, stream>>>(
        out_mu_l, out_lv_l, ABUF, Z, Zpad, total);
  }

  // ---- decoder hidden: z @ w3 + b3 -> BN -> lrelu (K=40 padded to 64) ----
  run_layer(w3, b3, bnd_g, bnd_b, Z, Zpad, H);        // hd bf16 in ABUF

  // ---- mu_x = lrelu(hd @ w4 + b4), logvar_x = hd @ w5 + b5 ----
  {
    dim3 tgrid(H / 32, N / 32);
    dim3 grid(N / 128, B / 128);
    wtcvt_tiled_kernel<<<tgrid, 256, 0, stream>>>(w4, WBUF, H, N, H);
    gemm_bf16_wmma<<<grid, 256, 0, stream>>>(
        ABUF, WBUF, b4, out_mu_x, N, H, H, H, N, 1);
    wtcvt_tiled_kernel<<<tgrid, 256, 0, stream>>>(w5, WBUF, H, N, H);
    gemm_bf16_wmma<<<grid, 256, 0, stream>>>(
        ABUF, WBUF, b5, out_lv_x, N, H, H, H, N, 0);
  }
}
